// SequenceLSTM_52235392254157
// MI455X (gfx1250) — compile-verified, matrix operand order checked by reading, not run on hardware
//
#include <hip/hip_runtime.h>

typedef __attribute__((ext_vector_type(16))) _Float16 v16h;
typedef __attribute__((ext_vector_type(8)))  float    v8f;

#define T_STEPS  512
#define BATCH    256
#define FEAT     64
#define HID      256
#define NWG      16
#define NTHREADS 256
#define XELEMS   ((size_t)T_STEPS * BATCH * FEAT)   // 8388608

__device__ __forceinline__ float sigmoidf_(float x) {
  return 1.0f / (1.0f + __expf(-x));
}
// overflow-safe fast tanh: (1-e)/(1+e) on |x|, restore sign; e in (0,1]
__device__ __forceinline__ float tanh_fast(float x) {
  float ax = fabsf(x);
  float e  = __expf(-2.0f * ax);
  float t  = (1.0f - e) / (1.0f + e);
  return copysignf(t, x);
}

// A-fragment (16x32, f16) per documented CDNA5 layout:
// lane<16 : M=lane,    K in {k0..k0+7, k0+16..k0+23}
// lane>=16: M=lane-16, K in {k0+8..k0+15, k0+24..k0+31}
__device__ __forceinline__ v16h load_a_h16(const _Float16* __restrict__ src, int ld,
                                           int row0, int k0, int lane) {
  int hi = lane >> 4;
  int r  = row0 + (lane & 15);
  int ka = k0 + hi * 8;
  union { uint4 u[2]; v16h v; } t;
  t.u[0] = *(const uint4*)(src + r * ld + ka);
  t.u[1] = *(const uint4*)(src + r * ld + ka + 16);
  return t.v;
}

// Same fragment from an fp32 source (fallback when workspace too small for x16).
__device__ __forceinline__ v16h load_a_f32(const float* __restrict__ src, int ld,
                                           int row0, int k0, int lane) {
  int hi = lane >> 4;
  int r  = row0 + (lane & 15);
  int ka = k0 + hi * 8;
  const float4* p0 = (const float4*)(src + r * ld + ka);
  const float4* p1 = (const float4*)(src + r * ld + ka + 16);
  float4 f0 = p0[0], f1 = p0[1];
  float4 f2 = p1[0], f3 = p1[1];
  union { _Float16 h[16]; v16h v; } t;
  t.h[0]=(_Float16)f0.x;  t.h[1]=(_Float16)f0.y;  t.h[2]=(_Float16)f0.z;  t.h[3]=(_Float16)f0.w;
  t.h[4]=(_Float16)f1.x;  t.h[5]=(_Float16)f1.y;  t.h[6]=(_Float16)f1.z;  t.h[7]=(_Float16)f1.w;
  t.h[8]=(_Float16)f2.x;  t.h[9]=(_Float16)f2.y;  t.h[10]=(_Float16)f2.z; t.h[11]=(_Float16)f2.w;
  t.h[12]=(_Float16)f3.x; t.h[13]=(_Float16)f3.y; t.h[14]=(_Float16)f3.z; t.h[15]=(_Float16)f3.w;
  return t.v;
}

__global__ void lstm_init(_Float16* h_buf, int* cnt, int* gen) {
  int idx = blockIdx.x * blockDim.x + threadIdx.x;
  uint4 z = {0u, 0u, 0u, 0u};
  if (idx < (2 * BATCH * HID) / 8)          // zero both h double-buffers (f16)
    ((uint4*)h_buf)[idx] = z;
  if (idx == 0) { *cnt = 0; *gen = 0; }
}

// one-time bandwidth-bound fp32 -> f16 conversion of the whole x tensor
__global__ void lstm_xcvt(const float* __restrict__ x, _Float16* __restrict__ x16) {
  size_t i = ((size_t)blockIdx.x * blockDim.x + threadIdx.x) * 4;
  if (i >= XELEMS) return;
  float4 f = *(const float4*)(x + i);
  union { _Float16 h[4]; uint2 u; } t;
  t.h[0] = (_Float16)f.x; t.h[1] = (_Float16)f.y;
  t.h[2] = (_Float16)f.z; t.h[3] = (_Float16)f.w;
  *(uint2*)(x16 + i) = t.u;
}

__global__ __launch_bounds__(NTHREADS) void lstm_persistent(
    const float* __restrict__ x,     const _Float16* __restrict__ x16,
    int use_x16,
    const float* __restrict__ W_ih,  const float* __restrict__ W_hh,
    const float* __restrict__ b_ih,  const float* __restrict__ b_hh,
    const float* __restrict__ W_lin, const float* __restrict__ b_lin,
    float* __restrict__ y,
    _Float16* __restrict__ h_buf,    int* cnt, int* gen)
{
  // Pre-swizzled WMMA B-fragments (K x N = 32x16, f16), 2x uint4 per lane.
  __shared__ uint4 sB_hh[4][8][32][2];   // [gate][k-chunk of 256][lane][2]  = 32 KB
  __shared__ uint4 sB_ih[4][2][32][2];   // [gate][k-chunk of 64 ][lane][2]  =  8 KB

  const int wg   = blockIdx.x;           // owns h-columns [16*wg, 16*wg+16)
  const int tid  = threadIdx.x;
  const int wave = tid >> 5;
  const int lane = tid & 31;
  const int hi   = lane >> 4;
  const int nl   = lane & 15;            // N column within tile (C/D layout)

  // ---- one-time: gather W_hh / W_ih rows for our gate columns into LDS fragments
  for (int f = wave; f < 32; f += 8) {
    int g = f >> 3, kt = f & 7;
    int ncol = g * HID + wg * 16 + nl;                 // row of W_hh [4H, H]
    union { _Float16 h[16]; uint4 u[2]; } t;
    #pragma unroll
    for (int j = 0; j < 16; ++j) {
      int k = kt * 32 + hi * 8 + (j < 8 ? j : j + 8);  // mirror of A-fragment K map
      t.h[j] = (_Float16)W_hh[ncol * HID + k];
    }
    sB_hh[g][kt][lane][0] = t.u[0];
    sB_hh[g][kt][lane][1] = t.u[1];
  }
  for (int f = wave; f < 8; f += 8) {                  // one fragment per wave
    int g = f >> 1, kt = f & 1;
    int ncol = g * HID + wg * 16 + nl;                 // row of W_ih [4H, F]
    union { _Float16 h[16]; uint4 u[2]; } t;
    #pragma unroll
    for (int j = 0; j < 16; ++j) {
      int k = kt * 32 + hi * 8 + (j < 8 ? j : j + 8);
      t.h[j] = (_Float16)W_ih[ncol * FEAT + k];
    }
    sB_ih[g][kt][lane][0] = t.u[0];
    sB_ih[g][kt][lane][1] = t.u[1];
  }

  // per-lane gate biases (depend only on the N column of the tile)
  float bias[4];
  #pragma unroll
  for (int g = 0; g < 4; ++g) {
    int n = g * HID + wg * 16 + nl;
    bias[g] = b_ih[n] + b_hh[n];
  }

  // cell state lives in registers for the whole sequence
  float creg[2][8];
  #pragma unroll
  for (int m = 0; m < 2; ++m)
    #pragma unroll
    for (int r = 0; r < 8; ++r) creg[m][r] = 0.0f;

  __syncthreads();

  _Float16* hb0 = h_buf;
  _Float16* hb1 = h_buf + BATCH * HID;
  const int row0a = (2 * wave) * 16;     // this wave's two 16-row batch tiles
  const int row0b = row0a + 16;

  for (int t = 0; t < T_STEPS; ++t) {
    const _Float16* h_src = (t & 1) ? hb1 : hb0;
    _Float16*       h_dst = (t & 1) ? hb0 : hb1;

    v8f acc[2][4];
    #pragma unroll
    for (int mt = 0; mt < 2; ++mt)
      #pragma unroll
      for (int g = 0; g < 4; ++g) {
        v8f z;
        #pragma unroll
        for (int r = 0; r < 8; ++r) z[r] = bias[g];
        acc[mt][g] = z;
      }

    // ---- input contribution: x_t (B x 64) * W_ih^T slice (K-chunk outer, mt inner)
    if (use_x16) {
      const _Float16* x_t = x16 + (size_t)t * BATCH * FEAT;
      #pragma unroll
      for (int kt = 0; kt < 2; ++kt) {
        union { uint4 u[2]; v16h v; } bm[4];
        #pragma unroll
        for (int g = 0; g < 4; ++g) {
          bm[g].u[0] = sB_ih[g][kt][lane][0];
          bm[g].u[1] = sB_ih[g][kt][lane][1];
        }
        v16h a0 = load_a_h16(x_t, FEAT, row0a, kt * 32, lane);
        v16h a1 = load_a_h16(x_t, FEAT, row0b, kt * 32, lane);
        #pragma unroll
        for (int g = 0; g < 4; ++g)
          acc[0][g] = __builtin_amdgcn_wmma_f32_16x16x32_f16(
              false, a0, false, bm[g].v, (short)0, acc[0][g], false, false);
        #pragma unroll
        for (int g = 0; g < 4; ++g)
          acc[1][g] = __builtin_amdgcn_wmma_f32_16x16x32_f16(
              false, a1, false, bm[g].v, (short)0, acc[1][g], false, false);
      }
    } else {
      const float* x_t = x + (size_t)t * BATCH * FEAT;
      #pragma unroll
      for (int kt = 0; kt < 2; ++kt) {
        union { uint4 u[2]; v16h v; } bm[4];
        #pragma unroll
        for (int g = 0; g < 4; ++g) {
          bm[g].u[0] = sB_ih[g][kt][lane][0];
          bm[g].u[1] = sB_ih[g][kt][lane][1];
        }
        v16h a0 = load_a_f32(x_t, FEAT, row0a, kt * 32, lane);
        v16h a1 = load_a_f32(x_t, FEAT, row0b, kt * 32, lane);
        #pragma unroll
        for (int g = 0; g < 4; ++g)
          acc[0][g] = __builtin_amdgcn_wmma_f32_16x16x32_f16(
              false, a0, false, bm[g].v, (short)0, acc[0][g], false, false);
        #pragma unroll
        for (int g = 0; g < 4; ++g)
          acc[1][g] = __builtin_amdgcn_wmma_f32_16x16x32_f16(
              false, a1, false, bm[g].v, (short)0, acc[1][g], false, false);
      }
    }

    // ---- recurrent contribution: h (B x 256) * W_hh^T slice
    #pragma unroll
    for (int kt = 0; kt < 8; ++kt) {
      union { uint4 u[2]; v16h v; } bm[4];
      #pragma unroll
      for (int g = 0; g < 4; ++g) {
        bm[g].u[0] = sB_hh[g][kt][lane][0];
        bm[g].u[1] = sB_hh[g][kt][lane][1];
      }
      v16h a0 = load_a_h16(h_src, HID, row0a, kt * 32, lane);
      v16h a1 = load_a_h16(h_src, HID, row0b, kt * 32, lane);
      #pragma unroll
      for (int g = 0; g < 4; ++g)
        acc[0][g] = __builtin_amdgcn_wmma_f32_16x16x32_f16(
            false, a0, false, bm[g].v, (short)0, acc[0][g], false, false);
      #pragma unroll
      for (int g = 0; g < 4; ++g)
        acc[1][g] = __builtin_amdgcn_wmma_f32_16x16x32_f16(
            false, a1, false, bm[g].v, (short)0, acc[1][g], false, false);
    }

    // ---- LSTM cell update — i/f/g/o for the same (b,n) sit in the same lane/VGPR
    #pragma unroll
    for (int mt = 0; mt < 2; ++mt) {
      const int row0 = mt ? row0b : row0a;
      #pragma unroll
      for (int r = 0; r < 8; ++r) {
        float iv = sigmoidf_(acc[mt][0][r]);
        float fv = sigmoidf_(acc[mt][1][r]);
        float gv = tanh_fast(acc[mt][2][r]);
        float ov = sigmoidf_(acc[mt][3][r]);
        float cn = fv * creg[mt][r] + iv * gv;
        creg[mt][r] = cn;
        float hn = ov * tanh_fast(cn);
        int b = row0 + r + hi * 8;                     // C/D layout row
        h_dst[b * HID + wg * 16 + nl] = (_Float16)hn;
      }
    }

    // prefetch next timestep's x slab while we sync
    if (t + 1 < T_STEPS) {
      if (use_x16)
        __builtin_prefetch(x16 + (size_t)(t + 1) * BATCH * FEAT + tid * 128, 0, 0);
      else
        __builtin_prefetch(x + (size_t)(t + 1) * BATCH * FEAT + tid * 64, 0, 0);
    }

    // device-wide split barrier (16 co-resident workgroups)
    __threadfence();
    __syncthreads();
    if (tid == 0) {
      int g = *(volatile int*)gen;
      if (atomicAdd(cnt, 1) == NWG - 1) {
        atomicExch(cnt, 0);
        __threadfence();
        *(volatile int*)gen = g + 1;
      } else {
        while (*(volatile int*)gen == g) __builtin_amdgcn_s_sleep(1);
      }
      __threadfence();
    }
    __syncthreads();
  }

  // head: y[b] = h_last[b,:] . W_lin + b_lin   (final h is in buffer 0)
  if (wg == 0) {
    const _Float16* hf = hb0;
    int b = tid;
    float s = 0.0f;
    #pragma unroll 8
    for (int k = 0; k < HID; ++k) s += (float)hf[b * HID + k] * W_lin[k];
    y[b] = s + b_lin[0];
  }
}

extern "C" void kernel_launch(void* const* d_in, const int* in_sizes, int n_in,
                              void* d_out, int out_size, void* d_ws, size_t ws_size,
                              hipStream_t stream) {
  const float* x     = (const float*)d_in[0];
  const float* W_ih  = (const float*)d_in[1];
  const float* W_hh  = (const float*)d_in[2];
  const float* b_ih  = (const float*)d_in[3];
  const float* b_hh  = (const float*)d_in[4];
  const float* W_lin = (const float*)d_in[5];
  const float* b_lin = (const float*)d_in[6];
  float* y = (float*)d_out;

  const size_t h_bytes = (size_t)2 * BATCH * HID * sizeof(_Float16);   // 256 KB
  _Float16* h_buf = (_Float16*)d_ws;
  int*      cnt   = (int*)((char*)d_ws + h_bytes);
  int*      gen   = cnt + 1;
  _Float16* x16   = (_Float16*)((char*)d_ws + h_bytes + 256);

  const size_t need_x16 = h_bytes + 256 + XELEMS * sizeof(_Float16);   // ~17 MB
  const int use_x16 = (ws_size >= need_x16) ? 1 : 0;

  lstm_init<<<64, 256, 0, stream>>>(h_buf, cnt, gen);
  if (use_x16) {
    int nthreads4 = (int)(XELEMS / 4);                 // 2M threads, float4 each
    lstm_xcvt<<<nthreads4 / 256, 256, 0, stream>>>(x, x16);
  }
  lstm_persistent<<<NWG, NTHREADS, 0, stream>>>(x, x16, use_x16,
                                                W_ih, W_hh, b_ih, b_hh,
                                                W_lin, b_lin, y, h_buf, cnt, gen);
}
